// Memory_9208409882686
// MI455X (gfx1250) — matrix operand order repaired and checked by reading, not compiled, not run on hardware
//
#include <hip/hip_runtime.h>

// Problem constants (from reference)
#define N_Q  16384   // B*H*W
#define DIM  512
#define M_K  2000
#define INV_T 10.0f  // 1/temp for both softmaxes

typedef __attribute__((ext_vector_type(16))) _Float16 v16h;
typedef __attribute__((ext_vector_type(8)))  _Float16 v8h;
typedef __attribute__((ext_vector_type(8)))  float    v8f;

// ---------------------------------------------------------------------------
// K1: l2-normalize query rows; emit qf (f32), qf_h (f16), and first half of
//     updated_query (channel-major scatter).
// ---------------------------------------------------------------------------
__global__ void qnorm_kernel(const float* __restrict__ query,
                             float* __restrict__ qf,
                             _Float16* __restrict__ qf_h,
                             float* __restrict__ uq)
{
    const int n   = blockIdx.x;          // 0..N_Q-1
    const int b   = n >> 10;             // H*W = 1024
    const int hw  = n & 1023;
    const int tid = threadIdx.x;
    __shared__ float red[256];

    const float* src = query + (size_t)b * DIM * 1024 + hw;  // stride 1024 over d
    float ss = 0.f;
    for (int d = tid; d < DIM; d += 256) { float v = src[(size_t)d * 1024]; ss += v * v; }
    red[tid] = ss; __syncthreads();
    for (int s = 128; s > 0; s >>= 1) { if (tid < s) red[tid] += red[tid + s]; __syncthreads(); }
    const float scale = 1.f / fmaxf(sqrtf(red[0]), 1e-12f);

    for (int d = tid; d < DIM; d += 256) {
        float v = src[(size_t)d * 1024] * scale;
        qf  [(size_t)n * DIM + d] = v;
        qf_h[(size_t)n * DIM + d] = (_Float16)v;
        uq[((size_t)b * 1024 + d) * 1024 + hw] = v;   // updated_query[:, 0:512, :, :]
    }
}

// ---------------------------------------------------------------------------
// K2: key prep — normalized f16 copy (for score GEMM) and K-major f16
//     transpose of the raw keys (for the concat GEMM's B fragments).
// ---------------------------------------------------------------------------
__global__ void keys_prep(const float* __restrict__ keys,
                          _Float16* __restrict__ keysn_h,
                          _Float16* __restrict__ keysT_h)
{
    const int j = blockIdx.x, tid = threadIdx.x;
    __shared__ float red[256];
    float ss = 0.f;
    for (int d = tid; d < DIM; d += 256) { float v = keys[(size_t)j * DIM + d]; ss += v * v; }
    red[tid] = ss; __syncthreads();
    for (int s = 128; s > 0; s >>= 1) { if (tid < s) red[tid] += red[tid + s]; __syncthreads(); }
    const float scale = 1.f / fmaxf(sqrtf(red[0]), 1e-12f);
    for (int d = tid; d < DIM; d += 256) {
        float v = keys[(size_t)j * DIM + d];
        keysn_h[(size_t)j * DIM + d]  = (_Float16)(v * scale);
        keysT_h[(size_t)d * M_K + j]  = (_Float16)v;     // 512 x 2000, K-major
    }
}

// A-fragment loader (16x32 f16): halves 0..7 = K kb..kb+7, halves 8..15 = K 16+kb..
__device__ __forceinline__ v16h load_a16(const _Float16* __restrict__ p)
{
    v8h lo = *(const v8h*)(p);
    v8h hi = *(const v8h*)(p + 16);
    v16h a;
#pragma unroll
    for (int e = 0; e < 8; ++e) { a[e] = lo[e]; a[e + 8] = hi[e]; }
    return a;
}

// ---------------------------------------------------------------------------
// K3: GEMM1  score = qf_n (16384x512) * keys_n^T (512x2000), f16 WMMA, f32 acc.
//     Wave tile: 64 rows x 16 cols (B fragment reused 4x), software-pipelined
//     so the k+1 fragment loads are in flight during the 4 WMMAs of step k.
//     grid = (N_Q/64, 16), block = 256 (8 waves); col tiles 0..124 guarded.
// ---------------------------------------------------------------------------
__global__ void gemm1_score(const _Float16* __restrict__ qf_h,
                            const _Float16* __restrict__ keysn_h,
                            float* __restrict__ score)
{
    const int wave = threadIdx.x >> 5;
    const int lane = threadIdx.x & 31;
    const int rowBase = blockIdx.x * 64;
    const int colTile = blockIdx.y * 8 + wave;
    if (colTile >= (M_K / 16)) return;          // 125 col tiles
    const int colBase = colTile * 16;

    const int mn  = lane & 15;
    const int kbA = (lane < 16) ? 0 : 8;
    const int kbB = (lane < 16) ? 0 : 8;

    v8f acc[4] = {};
    const _Float16* brow  = keysn_h + (size_t)(colBase + mn) * DIM;
    const _Float16* arow0 = qf_h    + (size_t)(rowBase + mn) * DIM;

    // prologue: fragments for k0 = 0
    v16h bC = load_a16(brow + kbB);
    v16h aC[4];
#pragma unroll
    for (int t = 0; t < 4; ++t) aC[t] = load_a16(arow0 + (size_t)(16 * t) * DIM + kbA);

#pragma unroll 4
    for (int k0 = 0; k0 < DIM; k0 += 32) {
        const int k1 = k0 + 32;
        v16h bN = {};
        v16h aN[4] = {};
        if (k1 < DIM) {                       // issue next-step loads first
            bN = load_a16(brow + k1 + kbB);
#pragma unroll
            for (int t = 0; t < 4; ++t)
                aN[t] = load_a16(arow0 + (size_t)(16 * t) * DIM + k1 + kbA);
        }
#pragma unroll
        for (int t = 0; t < 4; ++t)
            acc[t] = __builtin_amdgcn_wmma_f32_16x16x32_f16(
                false, aC[t], false, bC, (short)0, acc[t], false, false);
        bC = bN;
#pragma unroll
        for (int t = 0; t < 4; ++t) aC[t] = aN[t];
    }

    const int rOff = (lane < 16) ? 0 : 8;
#pragma unroll
    for (int t = 0; t < 4; ++t) {
#pragma unroll
        for (int r = 0; r < 8; ++r)
            score[(size_t)(rowBase + 16 * t + r + rOff) * M_K + colBase + mn] = acc[t][r];
    }
}

// ---------------------------------------------------------------------------
// K4: fused per-row: top-2 (+indices), row softmax -> score_memory, and both
//     loss contributions (compactness, separateness) via atomic accumulators.
// ---------------------------------------------------------------------------
__global__ void row_softmax(const float* __restrict__ raw,
                            const float* __restrict__ qf,
                            const float* __restrict__ keys,
                            float* __restrict__ sm,
                            int* __restrict__ g1, int* __restrict__ g2,
                            float* __restrict__ s_top,
                            float* __restrict__ lossa)
{
    const int row = blockIdx.x, tid = threadIdx.x;
    __shared__ float sv1[256]; __shared__ int si1[256];
    __shared__ float sv2[256]; __shared__ int si2[256];
    __shared__ float red[256];

    const float* r = raw + (size_t)row * M_K;
    float m1 = -1e30f, m2 = -1e30f; int i1 = 0, i2 = 0;
    for (int j = tid; j < M_K; j += 256) {
        float v = r[j];
        if (v > m1)      { m2 = m1; i2 = i1; m1 = v; i1 = j; }
        else if (v > m2) { m2 = v; i2 = j; }
    }
    sv1[tid] = m1; si1[tid] = i1; sv2[tid] = m2; si2[tid] = i2; __syncthreads();
    for (int s = 128; s > 0; s >>= 1) {
        if (tid < s) {
            float bm1 = sv1[tid + s]; int bi1 = si1[tid + s];
            float bm2 = sv2[tid + s]; int bi2 = si2[tid + s];
            float am1 = sv1[tid];     int ai1 = si1[tid];
            float am2 = sv2[tid];     int ai2 = si2[tid];
            float c1, c2; int j1, j2;
            if (bm1 > am1) { c1 = bm1; j1 = bi1;
                if (am1 > bm2) { c2 = am1; j2 = ai1; } else { c2 = bm2; j2 = bi2; } }
            else           { c1 = am1; j1 = ai1;
                if (bm1 > am2) { c2 = bm1; j2 = bi1; } else { c2 = am2; j2 = ai2; } }
            sv1[tid] = c1; si1[tid] = j1; sv2[tid] = c2; si2[tid] = j2;
        }
        __syncthreads();
    }
    const float rm = sv1[0]; const int top1 = si1[0], topn = si2[0];

    float se = 0.f;
    for (int j = tid; j < M_K; j += 256) se += __expf((r[j] - rm) * INV_T);
    red[tid] = se; __syncthreads();
    for (int s = 128; s > 0; s >>= 1) { if (tid < s) red[tid] += red[tid + s]; __syncthreads(); }
    const float inv = 1.f / red[0];
    for (int j = tid; j < M_K; j += 256)
        sm[(size_t)row * M_K + j] = __expf((r[j] - rm) * INV_T) * inv;
    if (tid == 0) { g1[row] = top1; g2[row] = topn; s_top[row] = rm; }

    // distances to pos/neg keys
    float comp = 0.f, dp2 = 0.f, dn2 = 0.f;
    const float* q  = qf   + (size_t)row  * DIM;
    const float* kp = keys + (size_t)top1 * DIM;
    const float* kn = keys + (size_t)topn * DIM;
    for (int d = tid; d < DIM; d += 256) {
        float qv = q[d], dP = qv - kp[d], dN = qv - kn[d];
        comp += dP * dP;
        float a = dP + 1e-6f, bb = dN + 1e-6f;
        dp2 += a * a; dn2 += bb * bb;
    }
    __syncthreads();
    red[tid] = comp; __syncthreads();
    for (int s = 128; s > 0; s >>= 1) { if (tid < s) red[tid] += red[tid + s]; __syncthreads(); }
    float compSum = red[0]; __syncthreads();
    red[tid] = dp2; __syncthreads();
    for (int s = 128; s > 0; s >>= 1) { if (tid < s) red[tid] += red[tid + s]; __syncthreads(); }
    float dpSum = red[0]; __syncthreads();
    red[tid] = dn2; __syncthreads();
    for (int s = 128; s > 0; s >>= 1) { if (tid < s) red[tid] += red[tid + s]; __syncthreads(); }
    if (tid == 0) {
        atomicAdd(&lossa[0], compSum);
        atomicAdd(&lossa[1], fmaxf(sqrtf(dpSum) - sqrtf(red[0]) + 1.f, 0.f));
    }
}

// ---------------------------------------------------------------------------
// K5: column softmax (score_query, in place) — online max/sum partials,
//     combine, normalize. Column-parallel threads -> fully coalesced.
// ---------------------------------------------------------------------------
__global__ void col_softmax_p1(const float* __restrict__ sq,
                               float* __restrict__ partM, float* __restrict__ partS)
{
    const int col = blockIdx.x * 256 + threadIdx.x;
    const int chunk = blockIdx.y;            // 16 chunks of 1024 rows
    if (col >= M_K) return;
    float m = -1e30f, s = 0.f;
    const int r0 = chunk * 1024;
    for (int r = r0; r < r0 + 1024; ++r) {
        float v = sq[(size_t)r * M_K + col];
        if (v > m) { s = s * __expf((m - v) * INV_T) + 1.f; m = v; }
        else       { s += __expf((v - m) * INV_T); }
    }
    partM[chunk * 2048 + col] = m;
    partS[chunk * 2048 + col] = s;
}

__global__ void col_softmax_p2(const float* __restrict__ partM,
                               const float* __restrict__ partS,
                               float* __restrict__ colM, float* __restrict__ colS)
{
    const int col = blockIdx.x * 256 + threadIdx.x;
    if (col >= M_K) return;
    float m = -1e30f;
#pragma unroll
    for (int c = 0; c < 16; ++c) m = fmaxf(m, partM[c * 2048 + col]);
    float s = 0.f;
#pragma unroll
    for (int c = 0; c < 16; ++c) s += partS[c * 2048 + col] * __expf((partM[c * 2048 + col] - m) * INV_T);
    colM[col] = m; colS[col] = s;
}

__global__ void col_softmax_p3(float* __restrict__ sq,
                               const float* __restrict__ colM,
                               const float* __restrict__ colS)
{
    const int col = blockIdx.x * 256 + threadIdx.x;
    const int chunk = blockIdx.y;
    if (col >= M_K) return;
    const float m = colM[col], inv = 1.f / colS[col];
    const int r0 = chunk * 1024;
    for (int r = r0; r < r0 + 1024; ++r) {
        size_t idx = (size_t)r * M_K + col;
        sq[idx] = __expf((sq[idx] - m) * INV_T) * inv;
    }
}

// ---------------------------------------------------------------------------
// K6/K7: segment-weight computation and scatter into query_update.
//   col_max(score_query[:,g]) == exp(0)/colSum == 1/colSum (max is monotone).
// ---------------------------------------------------------------------------
__global__ void wraw_kernel(const int* __restrict__ g1, const float* __restrict__ s_top,
                            const float* __restrict__ colM, const float* __restrict__ colS,
                            float* __restrict__ w_raw, float* __restrict__ denom)
{
    const int i = blockIdx.x * 256 + threadIdx.x;
    if (i >= N_Q) return;
    const int gi = g1[i];
    const float S  = colS[gi];
    const float sv = __expf((s_top[i] - colM[gi]) * INV_T) / S;  // score_query[i, g[i]]
    const float cm = 1.f / S;                                     // col_max[g[i]]
    const float wv = sv / (cm + 1e-8f);
    w_raw[i] = wv;
    atomicAdd(&denom[gi], wv);
}

__global__ void scatter_kernel(const int* __restrict__ g1, const float* __restrict__ w_raw,
                               const float* __restrict__ denom, const float* __restrict__ qf,
                               float* __restrict__ qu)
{
    const int i = blockIdx.x, tid = threadIdx.x;
    const int gi = g1[i];
    const float wv = w_raw[i] / (denom[gi] + 1e-8f);
    for (int d = tid; d < DIM; d += 256)
        atomicAdd(&qu[(size_t)gi * DIM + d], wv * qf[(size_t)i * DIM + d]);
}

// ---------------------------------------------------------------------------
// K8: updated_memory = l2norm(query_update + keys)
// ---------------------------------------------------------------------------
__global__ void upd_mem_kernel(const float* __restrict__ qu, const float* __restrict__ keys,
                               float* __restrict__ um)
{
    const int j = blockIdx.x, tid = threadIdx.x;
    __shared__ float red[256];
    float ss = 0.f;
    for (int d = tid; d < DIM; d += 256) {
        float v = qu[(size_t)j * DIM + d] + keys[(size_t)j * DIM + d];
        ss += v * v;
    }
    red[tid] = ss; __syncthreads();
    for (int s = 128; s > 0; s >>= 1) { if (tid < s) red[tid] += red[tid + s]; __syncthreads(); }
    const float scale = 1.f / fmaxf(sqrtf(red[0]), 1e-12f);
    for (int d = tid; d < DIM; d += 256) {
        float v = qu[(size_t)j * DIM + d] + keys[(size_t)j * DIM + d];
        um[(size_t)j * DIM + d] = v * scale;
    }
}

// ---------------------------------------------------------------------------
// K9: GEMM2  concat = score_memory (16384x2000) * keys (2000x512), K tail
//     zero-padded, software-pipelined like GEMM1. Wave tile 16 rows x 64 cols
//     (A fragment reused 4x). Output scattered into updated_query[:,512:,:,:].
// ---------------------------------------------------------------------------
__global__ void gemm2_concat(const float* __restrict__ smem,
                             const _Float16* __restrict__ keysT_h,
                             float* __restrict__ uq)
{
    const int wave = threadIdx.x >> 5;
    const int lane = threadIdx.x & 31;
    const int rowBase = blockIdx.x * 128 + wave * 16;
    const int colBase = blockIdx.y * 64;
    const int mn  = lane & 15;
    const int kbA = (lane < 16) ? 0 : 8;
    const int kbB = (lane < 16) ? 0 : 8;

    v8f acc[4] = {};
    const float* arow = smem + (size_t)(rowBase + mn) * M_K;
    const _Float16* brow[4];
#pragma unroll
    for (int t = 0; t < 4; ++t) brow[t] = keysT_h + (size_t)(colBase + 16 * t + mn) * M_K;

    // A fragment: 8 f32 loads + cvt for halves 0..7, 8 more (or zeros) for 8..15
    auto loadA = [&](int k0, bool tail) {
        v16h a;
        const float* pa = arow + k0 + kbA;
#pragma unroll
        for (int e = 0; e < 8; ++e) a[e] = (_Float16)pa[e];
        if (!tail) {
#pragma unroll
            for (int e = 0; e < 8; ++e) a[8 + e] = (_Float16)pa[16 + e];
        } else {
#pragma unroll
            for (int e = 0; e < 8; ++e) a[8 + e] = (_Float16)0.f;
        }
        return a;
    };
    auto loadB = [&](int t, int k0, bool tail) {
        if (!tail) return load_a16(brow[t] + k0 + kbB);
        v16h b = {};                           // K >= 2000 -> zero
        const _Float16* pb = brow[t] + k0 + kbB;
#pragma unroll
        for (int e = 0; e < 8; ++e) b[e] = pb[e];
        return b;
    };

    // prologue (k0 = 0 is never the tail: M_K > 32)
    v16h aC = loadA(0, false);
    v16h bC[4];
#pragma unroll
    for (int t = 0; t < 4; ++t) bC[t] = loadB(t, 0, false);

#pragma unroll 2
    for (int k0 = 0; k0 < M_K; k0 += 32) {     // 63 steps; last is tail
        const int k1 = k0 + 32;
        v16h aN = {};
        v16h bN[4] = {};
        if (k1 < M_K) {                        // prefetch next step first
            const bool tl = (k1 + 32 > M_K);   // k1 == 1984 -> tail
            aN = loadA(k1, tl);
#pragma unroll
            for (int t = 0; t < 4; ++t) bN[t] = loadB(t, k1, tl);
        }
#pragma unroll
        for (int t = 0; t < 4; ++t)
            acc[t] = __builtin_amdgcn_wmma_f32_16x16x32_f16(
                false, aC, false, bC[t], (short)0, acc[t], false, false);
        aC = aN;
#pragma unroll
        for (int t = 0; t < 4; ++t) bC[t] = bN[t];
    }

    const int rOff = (lane < 16) ? 0 : 8;
#pragma unroll
    for (int t = 0; t < 4; ++t) {
        const int ch = 512 + colBase + 16 * t + mn;        // second half of channels
#pragma unroll
        for (int r = 0; r < 8; ++r) {
            const int row = rowBase + r + rOff;
            const int b   = row >> 10;
            const int hw  = row & 1023;
            uq[((size_t)b * 1024 + ch) * 1024 + hw] = acc[t][r];
        }
    }
}

__global__ void finalize(const float* __restrict__ lossa,
                         float* __restrict__ out_sep, float* __restrict__ out_comp)
{
    if (threadIdx.x == 0) {
        out_comp[0] = lossa[0] / ((float)N_Q * (float)DIM);
        out_sep[0]  = lossa[1] / (float)N_Q;
    }
}

// ---------------------------------------------------------------------------
extern "C" void kernel_launch(void* const* d_in, const int* in_sizes, int n_in,
                              void* d_out, int out_size, void* d_ws, size_t ws_size,
                              hipStream_t stream)
{
    (void)in_sizes; (void)n_in; (void)out_size; (void)ws_size;
    const float* query = (const float*)d_in[0];   // (16,512,32,32)
    const float* keys  = (const float*)d_in[1];   // (2000,512)

    float* out     = (float*)d_out;
    float* out_uq  = out;                                   // (16,1024,32,32)
    float* out_um  = out + 16777216;                        // (2000,512)
    float* out_sq  = out + 17801216;                        // (16384,2000)  raw->score_query
    float* out_sm  = out + 50569216;                        // (16384,2000)  score_memory
    float* out_sep = out + 83337216;
    float* out_comp= out + 83337217;

    char* w = (char*)d_ws;
    size_t off = 0;
    auto alloc = [&](size_t bytes) {
        void* p = w + off; off = (off + bytes + 255) & ~(size_t)255; return p;
    };
    float*    qf      = (float*)   alloc((size_t)N_Q * DIM * 4);
    _Float16* qf_h    = (_Float16*)alloc((size_t)N_Q * DIM * 2);
    _Float16* keysn_h = (_Float16*)alloc((size_t)M_K * DIM * 2);
    _Float16* keysT_h = (_Float16*)alloc((size_t)DIM * M_K * 2);
    int*      g1      = (int*)     alloc((size_t)N_Q * 4);
    int*      g2      = (int*)     alloc((size_t)N_Q * 4);
    float*    s_top   = (float*)   alloc((size_t)N_Q * 4);
    float*    w_raw   = (float*)   alloc((size_t)N_Q * 4);
    float*    denom   = (float*)   alloc((size_t)M_K * 4);
    float*    colM    = (float*)   alloc((size_t)2048 * 4);
    float*    colS    = (float*)   alloc((size_t)2048 * 4);
    float*    partM   = (float*)   alloc((size_t)16 * 2048 * 4);
    float*    partS   = (float*)   alloc((size_t)16 * 2048 * 4);
    float*    qu      = (float*)   alloc((size_t)M_K * DIM * 4);
    float*    lossa   = (float*)   alloc(256);

    (void)hipMemsetAsync(denom, 0, (size_t)M_K * 4, stream);
    (void)hipMemsetAsync(qu,    0, (size_t)M_K * DIM * 4, stream);
    (void)hipMemsetAsync(lossa, 0, 8, stream);

    qnorm_kernel   <<<N_Q, 256, 0, stream>>>(query, qf, qf_h, out_uq);
    keys_prep      <<<M_K, 256, 0, stream>>>(keys, keysn_h, keysT_h);
    gemm1_score    <<<dim3(N_Q / 64, 16), 256, 0, stream>>>(qf_h, keysn_h, out_sq);
    row_softmax    <<<N_Q, 256, 0, stream>>>(out_sq, qf, keys, out_sm, g1, g2, s_top, lossa);
    col_softmax_p1 <<<dim3(8, 16), 256, 0, stream>>>(out_sq, partM, partS);
    col_softmax_p2 <<<8, 256, 0, stream>>>(partM, partS, colM, colS);
    col_softmax_p3 <<<dim3(8, 16), 256, 0, stream>>>(out_sq, colM, colS);
    wraw_kernel    <<<N_Q / 256, 256, 0, stream>>>(g1, s_top, colM, colS, w_raw, denom);
    scatter_kernel <<<N_Q, 256, 0, stream>>>(g1, w_raw, denom, qf, qu);
    upd_mem_kernel <<<M_K, 256, 0, stream>>>(qu, keys, out_um);
    gemm2_concat   <<<dim3(N_Q / 128, 8), 256, 0, stream>>>(out_sm, keysT_h, out_uq);
    finalize       <<<1, 32, 0, stream>>>(lossa, out_sep, out_comp);
}
